// TorchGAT_46995532153585
// MI455X (gfx1250) — matrix operand order repaired and checked by reading, not compiled, not run on hardware
//
#include <hip/hip_runtime.h>

typedef __attribute__((ext_vector_type(16))) _Float16 v16h;
typedef __attribute__((ext_vector_type(8)))  _Float16 v8h;
typedef __attribute__((ext_vector_type(8)))  float    v8f;

#define NEG_SLOPE 0.2f
#define SA_STRIDE 136   // 16x128 A tile, padded (halves)
#define SB_STRIDE 136   // 128x128 W^T tile, padded (halves)

// ---------------------------------------------------------------------------
// feat[N,128] = h[N,128] @ W[128,128]  via v_wmma_f32_16x16x32_f16
// block = 256 threads = 8 waves; block handles 16 rows, wave w handles cols
// [16w,16w+16). K=128 -> 4 chained WMMAs per wave.
// ---------------------------------------------------------------------------
__global__ __launch_bounds__(256)
void gat_gemm_wmma(const float* __restrict__ h, const float* __restrict__ W,
                   float* __restrict__ feat, int n_nodes) {
    __shared__ _Float16 sA[16 * SA_STRIDE];     // [row][k]
    __shared__ _Float16 sBt[128 * SB_STRIDE];   // [col][k]  (W transposed)

    const int tid      = threadIdx.x;
    const int rowBase  = blockIdx.x * 16;
    const bool fullTile = (rowBase + 16 <= n_nodes);   // uniform per block

    // Stage A tile: coalesced global read, f32->f16 convert into LDS.
    if (fullTile) {
        for (int i = tid; i < 16 * 128; i += 256) {
            int r = i >> 7, k = i & 127;
            sA[r * SA_STRIDE + k] = (_Float16)h[(size_t)(rowBase + r) * 128 + k];
        }
    } else {
        for (int i = tid; i < 16 * 128; i += 256) {
            int r = i >> 7, k = i & 127;
            int row = rowBase + r;
            float v = (row < n_nodes) ? h[(size_t)row * 128 + k] : 0.0f;
            sA[r * SA_STRIDE + k] = (_Float16)v;
        }
    }
    // Stage all of W transposed: coalesced global read, transposed LDS write.
    for (int i = tid; i < 128 * 128; i += 256) {
        int k = i >> 7, c = i & 127;
        sBt[c * SB_STRIDE + k] = (_Float16)W[i];
    }
    __syncthreads();

    const int wave     = tid >> 5;
    const int lane     = tid & 31;
    const int laneHalf = lane >> 4;   // 0: lanes 0-15, 1: lanes 16-31
    const int rlow     = lane & 15;
    const int colBase  = wave * 16;

    v8f c = {};
#pragma unroll
    for (int kk = 0; kk < 4; ++kk) {
        const int kb = kk * 32;
        // A fragment (ISA 16-bit A 16x32 layout): row = rlow,
        // halves 0..7  -> K = kb + laneHalf*8 + 0..7
        // halves 8..15 -> K = kb + 16 + laneHalf*8 + 0..7
        const _Float16* pa = &sA[rlow * SA_STRIDE + kb + (laneHalf << 3)];
        v8h a0 = *(const v8h*)pa;
        v8h a1 = *(const v8h*)(pa + 16);
        v16h a;
#pragma unroll
        for (int i = 0; i < 8; ++i) { a[i] = a0[i]; a[i + 8] = a1[i]; }

        // B fragment: col = colBase + rlow, same K mapping as A
        const _Float16* pb = &sBt[(colBase + rlow) * SB_STRIDE + kb + (laneHalf << 3)];
        v8h b0 = *(const v8h*)pb;
        v8h b1 = *(const v8h*)(pb + 16);
        v16h b;
#pragma unroll
        for (int i = 0; i < 8; ++i) { b[i] = b0[i]; b[i + 8] = b1[i]; }

        c = __builtin_amdgcn_wmma_f32_16x16x32_f16(
                /*neg_a=*/false, a, /*neg_b=*/false, b,
                /*c_mod=*/(short)0, c, /*reuse_a=*/false, /*reuse_b=*/false);
    }

    // C layout: VGPR r -> row = r + 8*laneHalf, col = rlow.
    float* outp = feat + (size_t)(rowBase + (laneHalf << 3)) * 128 + colBase + rlow;
    if (fullTile) {
        // straight-line, divergence-free store epilogue (N % 16 == 0 path)
#pragma unroll
        for (int r = 0; r < 8; ++r)
            outp[(size_t)r * 128] = c[r];
    } else {
#pragma unroll
        for (int r = 0; r < 8; ++r) {
            int row = rowBase + r + (laneHalf << 3);
            if (row < n_nodes)
                feat[(size_t)row * 128 + colBase + rlow] = c[r];
        }
    }
}

// ---------------------------------------------------------------------------
// el[n,h] = sum_d feat[n,h,d]*al[h,d];  er likewise. One thread per (n,h).
// ---------------------------------------------------------------------------
__global__ __launch_bounds__(256)
void gat_elr(const float* __restrict__ feat,
             const float* __restrict__ al, const float* __restrict__ ar,
             float* __restrict__ el, float* __restrict__ er, int n_nodes) {
    int i = blockIdx.x * blockDim.x + threadIdx.x;   // n*4 + h
    if (i >= n_nodes * 4) return;
    int hh = i & 3;
    const float* f = feat + (size_t)(i >> 2) * 128 + hh * 32;
    const float* a = al + hh * 32;
    const float* b = ar + hh * 32;
    float sl = 0.f, sr = 0.f;
#pragma unroll
    for (int d = 0; d < 32; ++d) { float v = f[d]; sl += v * a[d]; sr += v * b[d]; }
    el[i] = sl;
    er[i] = sr;
}

__device__ __forceinline__ int lower_bound_i32(const int* __restrict__ a, int n, int key) {
    int lo = 0, hi = n;
    while (lo < hi) { int mid = (lo + hi) >> 1; if (a[mid] < key) lo = mid + 1; else hi = mid; }
    return lo;
}

// ---------------------------------------------------------------------------
// Edge softmax + aggregation, one wave32 per destination node (dst is sorted
// -> contiguous segment per node, zero atomics). Lane l owns d=l for all 4
// heads. mode bit0: residual, bit1: ELU, bit2: write head-mean to [N,32].
// ---------------------------------------------------------------------------
__global__ __launch_bounds__(256)
void gat_aggregate(const float* __restrict__ feat,
                   const float* __restrict__ el, const float* __restrict__ er,
                   const int* __restrict__ dst_sorted, const int* __restrict__ src,
                   const float* __restrict__ h_res, float* __restrict__ out,
                   int n_nodes, int n_edges, int mode) {
    const int node = blockIdx.x * 8 + (threadIdx.x >> 5);
    if (node >= n_nodes) return;
    const int lane = threadIdx.x & 31;

    const int lo = lower_bound_i32(dst_sorted, n_edges, node);
    const int hi = lower_bound_i32(dst_sorted, n_edges, node + 1);

    float ern[4];
#pragma unroll
    for (int h = 0; h < 4; ++h) ern[h] = er[node * 4 + h];

    // Pass 1: per-head max over incoming edges (lanes split edges, then reduce)
    float m[4] = { -3.402823466e38f, -3.402823466e38f, -3.402823466e38f, -3.402823466e38f };
    for (int i = lo + lane; i < hi; i += 32) {
        int s = src[i];
#pragma unroll
        for (int h = 0; h < 4; ++h) {
            float e = el[s * 4 + h] + ern[h];
            e = e > 0.f ? e : NEG_SLOPE * e;
            m[h] = fmaxf(m[h], e);
        }
    }
#pragma unroll
    for (int off = 16; off > 0; off >>= 1)
#pragma unroll
        for (int h = 0; h < 4; ++h)
            m[h] = fmaxf(m[h], __shfl_xor(m[h], off, 32));

    // Pass 2: all lanes walk every edge; lane l accumulates feat[...,d=l].
    float sum[4] = {0.f, 0.f, 0.f, 0.f};
    float acc[4] = {0.f, 0.f, 0.f, 0.f};
    for (int i = lo; i < hi; ++i) {
        int s = src[i];
        const float* fr = feat + (size_t)s * 128;
#pragma unroll
        for (int h = 0; h < 4; ++h) {
            float e = el[s * 4 + h] + ern[h];          // broadcast loads
            e = e > 0.f ? e : NEG_SLOPE * e;
            float w = __expf(e - m[h]);
            sum[h] += w;
            acc[h] += w * fr[h * 32 + lane];            // coalesced 512B row
        }
    }

    float val[4];
#pragma unroll
    for (int h = 0; h < 4; ++h) {
        float a = (sum[h] > 0.f) ? acc[h] / sum[h] : 0.f;
        if (mode & 1) a += h_res[(size_t)node * 128 + h * 32 + lane];
        if (mode & 2) a = (a > 0.f) ? a : (__expf(a) - 1.f);   // ELU(alpha=1)
        val[h] = a;
    }

    if (mode & 4) {   // final layer: mean over heads -> [N, 32]
        out[(size_t)node * 32 + lane] = 0.25f * (val[0] + val[1] + val[2] + val[3]);
    } else {
#pragma unroll
        for (int h = 0; h < 4; ++h)
            out[(size_t)node * 128 + h * 32 + lane] = val[h];
    }
}

// ---------------------------------------------------------------------------
extern "C" void kernel_launch(void* const* d_in, const int* in_sizes, int n_in,
                              void* d_out, int out_size, void* d_ws, size_t ws_size,
                              hipStream_t stream) {
    const float* x   = (const float*)d_in[0];
    const int*   src = (const int*)d_in[1];
    const int*   dst = (const int*)d_in[2];
    const float* W0  = (const float*)d_in[3];
    const float* al0 = (const float*)d_in[4];
    const float* ar0 = (const float*)d_in[5];
    const float* W1  = (const float*)d_in[6];
    const float* al1 = (const float*)d_in[7];
    const float* ar1 = (const float*)d_in[8];
    const float* W2  = (const float*)d_in[9];
    const float* al2 = (const float*)d_in[10];
    const float* ar2 = (const float*)d_in[11];
    float* out = (float*)d_out;

    const int N = in_sizes[0] / 128;   // 50000
    const int E = in_sizes[1];         // 800000

    float* ws   = (float*)d_ws;
    size_t NF   = (size_t)N * 128;
    float* feat = ws;
    float* el   = feat + NF;
    float* er   = el + (size_t)N * 4;
    float* hA   = er + (size_t)N * 4;
    float* hB   = hA + NF;

    const int gemmBlocks = (N + 15) / 16;
    const int elrBlocks  = (N * 4 + 255) / 256;
    const int aggBlocks  = (N + 7) / 8;

    // ---- Layer 0: no residual, ELU
    gat_gemm_wmma<<<gemmBlocks, 256, 0, stream>>>(x, W0, feat, N);
    gat_elr<<<elrBlocks, 256, 0, stream>>>(feat, al0, ar0, el, er, N);
    gat_aggregate<<<aggBlocks, 256, 0, stream>>>(feat, el, er, dst, src, x, hA, N, E, /*mode=*/2);

    // ---- Layer 1: residual, ELU
    gat_gemm_wmma<<<gemmBlocks, 256, 0, stream>>>(hA, W1, feat, N);
    gat_elr<<<elrBlocks, 256, 0, stream>>>(feat, al1, ar1, el, er, N);
    gat_aggregate<<<aggBlocks, 256, 0, stream>>>(feat, el, er, dst, src, hA, hB, N, E, /*mode=*/3);

    // ---- Layer 2: residual, no act, head-mean to d_out [N,32]
    gat_gemm_wmma<<<gemmBlocks, 256, 0, stream>>>(hB, W2, feat, N);
    gat_elr<<<elrBlocks, 256, 0, stream>>>(feat, al2, ar2, el, er, N);
    gat_aggregate<<<aggBlocks, 256, 0, stream>>>(feat, el, er, dst, src, hB, out, N, E, /*mode=*/5);
}
